// PatchedLinear_23845658428105
// MI455X (gfx1250) — 8-bit matrix operand order repaired and checked by reading, compiled, not run on hardware
//
#include <hip/hip_runtime.h>
#include <stdint.h>

typedef __attribute__((ext_vector_type(16))) int   v16i;
typedef __attribute__((ext_vector_type(8)))  float v8f;
typedef int gv4i __attribute__((vector_size(16)));

#define FP8_MAX_V 448.0f

#define AS1 __attribute__((address_space(1)))
#define AS3 __attribute__((address_space(3)))

#if defined(__AMDGCN__) && __has_builtin(__builtin_amdgcn_global_load_async_to_lds_b128) && \
    __has_builtin(__builtin_amdgcn_s_wait_asynccnt)
#define HAVE_ASYNC_LDS 1
#else
#define HAVE_ASYNC_LDS 0
#endif

static __device__ __forceinline__ void sched_fence() {
#if defined(__AMDGCN__) && __has_builtin(__builtin_amdgcn_sched_barrier)
  __builtin_amdgcn_sched_barrier(0);
#endif
}

static __device__ __forceinline__ float clampf8(float v) {
  return fminf(fmaxf(v, -FP8_MAX_V), FP8_MAX_V);
}

static __device__ __forceinline__ uint32_t pack4_fp8(float a, float b, float c, float d) {
  int v = 0;
  v = __builtin_amdgcn_cvt_pk_fp8_f32(a, b, v, false);
  v = __builtin_amdgcn_cvt_pk_fp8_f32(c, d, v, true);
  return (uint32_t)v;
}

static __device__ __forceinline__ void copy16_g2l(const uint8_t* g, uint8_t* l) {
#if HAVE_ASYNC_LDS
  __builtin_amdgcn_global_load_async_to_lds_b128((AS1 gv4i*)g, (AS3 gv4i*)l, 0, 0);
#else
  *reinterpret_cast<uint4*>(l) = *reinterpret_cast<const uint4*>(g);
#endif
}

static __device__ __forceinline__ void wait_async_all() {
#if HAVE_ASYNC_LDS
  __builtin_amdgcn_s_wait_asynccnt(0);
#endif
}

__global__ void quant_x_kernel(const float* __restrict__ x,
                               const float* __restrict__ scale,
                               uint32_t* __restrict__ qx, int n4) {
  int i = blockIdx.x * blockDim.x + threadIdx.x;
  if (i >= n4) return;
  const float inv = 1.0f / scale[0];
  float4 v = reinterpret_cast<const float4*>(x)[i];
  qx[i] = pack4_fp8(clampf8(v.x * inv), clampf8(v.y * inv),
                    clampf8(v.z * inv), clampf8(v.w * inv));
}

__global__ __launch_bounds__(256)
void quant_wt_kernel(const float* __restrict__ w,
                     const float* __restrict__ scale,
                     uint8_t* __restrict__ qwt, int K, int N) {
  __shared__ uint8_t t[64][80];
  const float inv = 1.0f / scale[0];
  const int tid   = threadIdx.x;
  const int kbase = blockIdx.y * 64;
  const int nbase = blockIdx.x * 64;

  const int cn = tid & 63;
  const int r0 = (tid >> 6) << 4;
  #pragma unroll
  for (int j = 0; j < 16; j += 2) {
    int rk = r0 + j;
    float a = clampf8(w[(size_t)(kbase + rk + 0) * N + nbase + cn] * inv);
    float b = clampf8(w[(size_t)(kbase + rk + 1) * N + nbase + cn] * inv);
    int p = __builtin_amdgcn_cvt_pk_fp8_f32(a, b, 0, false);
    *reinterpret_cast<uint16_t*>(&t[cn][rk]) = (uint16_t)p;
  }
  __syncthreads();

  const int nr = tid >> 2;
  const int ch = (tid & 3) * 16;
  uint4 v = *reinterpret_cast<const uint4*>(&t[nr][ch]);
  *reinterpret_cast<uint4*>(qwt + (size_t)(nbase + nr) * K + kbase + ch) = v;
}

__global__ __launch_bounds__(256, 2)
void fp8_gemm_bias_kernel(const uint8_t* __restrict__ qx,
                          const uint8_t* __restrict__ qwt,
                          const float* __restrict__ bias,
                          const float* __restrict__ s_in,
                          const float* __restrict__ s_w,
                          float* __restrict__ out,
                          int M, int N, int K) {
  extern __shared__ uint8_t smem[];

  const int tid  = threadIdx.x;
  const int lane = tid & 31;
  const int wave = tid >> 5;
  const int half = lane >> 4;
  const int l16  = lane & 15;
  const int wm   = wave >> 2;
  const int wn   = wave & 3;

  const int bm = blockIdx.y * 128;
  const int bn = blockIdx.x * 128;

  const float dq = s_in[0] * s_w[0];

  const uint8_t* gA = qx  + (size_t)bm * K;
  const uint8_t* gB = qwt + (size_t)bn * K;

  v8f acc[4][2];
  #pragma unroll
  for (int mt = 0; mt < 4; ++mt)
    #pragma unroll
    for (int nt = 0; nt < 2; ++nt)
      #pragma unroll
      for (int r = 0; r < 8; ++r)
        acc[mt][nt][r] = 0.0f;

  auto stage = [&](int p, int k0) {
    uint8_t* sA = smem + p * 32768;
    uint8_t* sB = sA + 16384;
    #pragma unroll
    for (int r = 0; r < 4; ++r) {
      int li = r * 256 + tid;
      int rr = li >> 3;
      int kk = (li & 7) << 4;
      copy16_g2l(gA + (size_t)rr * K + k0 + kk, sA + li * 16);
      copy16_g2l(gB + (size_t)rr * K + k0 + kk, sB + li * 16);
    }
  };

  stage(0, 0);
  int pb = 0;

  for (int k0 = 0; k0 < K; k0 += 128) {
    wait_async_all();
    __syncthreads();
    if (k0 + 128 < K) stage(pb ^ 1, k0 + 128);

    const uint8_t* sA = smem + pb * 32768;
    const uint8_t* sB = sA + 16384;


    v16i Bf[2];
    #pragma unroll
    for (int nt = 0; nt < 2; ++nt) {
      const uint8_t* brow = sB + (size_t)(wn * 32 + nt * 16 + l16) * 128;
      #pragma unroll
      for (int c = 0; c < 4; ++c) {
        uint4 q = *reinterpret_cast<const uint4*>(brow + c * 32 + half * 16);
        Bf[nt][4 * c + 0] = (int)q.x;
        Bf[nt][4 * c + 1] = (int)q.y;
        Bf[nt][4 * c + 2] = (int)q.z;
        Bf[nt][4 * c + 3] = (int)q.w;
      }
    }

    v16i Af[4];
    #pragma unroll
    for (int mt = 0; mt < 4; ++mt) {
      const uint8_t* arow = sA + (size_t)(wm * 64 + mt * 16 + l16) * 128;
      #pragma unroll
      for (int hkb = 0; hkb < 2; ++hkb) {
        #pragma unroll
        for (int c = 0; c < 4; ++c) {
          uint2 q = *reinterpret_cast<const uint2*>(arow + hkb * 64 + (c >> 1) * 32 + half * 16 + (c & 1) * 8);
          Af[mt][8 * hkb + 2 * c + 0] = (int)q.x;
          Af[mt][8 * hkb + 2 * c + 1] = (int)q.y;
        }
      }
    }

    sched_fence();

    #pragma unroll
    for (int mt = 0; mt < 4; ++mt) {
      #pragma unroll
      for (int nt = 0; nt < 2; ++nt) {
        acc[mt][nt] = __builtin_amdgcn_wmma_f32_16x16x128_fp8_fp8(
            Af[mt], Bf[nt], (short)0, acc[mt][nt], false, false);
      }
    }
    pb ^= 1;
  }

  #pragma unroll
  for (int mt = 0; mt < 4; ++mt) {
    #pragma unroll
    for (int nt = 0; nt < 2; ++nt) {
      int gn = bn + wn * 32 + nt * 16 + l16;
      float bv = bias[gn];
      #pragma unroll
      for (int r = 0; r < 8; ++r) {
        int gm = bm + wm * 64 + mt * 16 + half * 8 + r;
        out[(size_t)gm * N + gn] = acc[mt][nt][r] * dq + bv;
      }
    }
  }
}

extern "C" void kernel_launch(void* const* d_in, const int* in_sizes, int n_in,
                              void* d_out, int out_size, void* d_ws, size_t ws_size,
                              hipStream_t stream) {
  const float* x    = (const float*)d_in[0];
  const float* w    = (const float*)d_in[1];
  const float* bias = (const float*)d_in[2];
  const float* s_in = (const float*)d_in[3];
  const float* s_w  = (const float*)d_in[4];
  float* out = (float*)d_out;

  const int N = in_sizes[2];
  const int K = in_sizes[1] / N;
  const int M = in_sizes[0] / K;

  uint8_t* qx  = (uint8_t*)d_ws;
  uint8_t* qwt = qx + (size_t)M * (size_t)K;

  {
    int n4 = (M * K) / 4;
    quant_x_kernel<<<(n4 + 255) / 256, 256, 0, stream>>>(x, s_in, (uint32_t*)qx, n4);
  }
  {
    dim3 grid(N / 64, K / 64);
    quant_wt_kernel<<<grid, 256, 0, stream>>>(w, s_w, qwt, K, N);
  }
  {
    dim3 grid(N / 128, M / 128);
    fp8_gemm_bias_kernel<<<grid, 256, 65536, stream>>>(qx, qwt, bias, s_in, s_w,
                                                       out, M, N, K);
  }
}
